// ExpanderPNA_23502061043981
// MI455X (gfx1250) — matrix operand order repaired and checked by reading, not compiled, not run on hardware
//
#include <hip/hip_runtime.h>
#include <hip/hip_bf16.h>

typedef _Float16 v16h __attribute__((ext_vector_type(16)));
typedef float    v8f  __attribute__((ext_vector_type(8)));

#define NN       30000
#define NE       70000
#define NG       1024
#define NTOW     5
#define FIN      75
#define FOUT     15
#define NL       4
#define MSG_K    225      // 3*FIN
#define MSG_KP   256      // padded K
#define MSG_N    375      // NTOW*FIN
#define MSG_NP   384      // padded N
#define MSG_NT   24       // MSG_NP/16
#define MSG_KK   8        // MSG_KP/32
#define POST_K   975      // 13*FIN
#define POST_KP  992
#define POST_KK  31
#define ADL      1.1824939f   // AVG_DEG_LOG

// ---------- helpers ----------
__device__ __forceinline__ unsigned fmap(float x) {
    unsigned b = __float_as_uint(x);
    return (b & 0x80000000u) ? ~b : (b | 0x80000000u);
}
__device__ __forceinline__ float funmap(unsigned u) {
    unsigned b = (u & 0x80000000u) ? (u & 0x7FFFFFFFu) : ~u;
    return __uint_as_float(b);
}
// fragment-order slot for A element (row r in 0..15, k): contiguous k = 16*(lane/16) + j, same order as the B packing
__device__ __forceinline__ int a_frag_off(int r, int k) {
    int kk = k >> 5, krem = k & 31;
    int lane = r + ((krem & 16) ? 16 : 0);
    int j = krem & 15;
    return ((kk * 32 + lane) << 4) + j;
}

// ---------- weight prep ----------
// enc_tab[l][a][o] = edge_emb[a] @ enc_w[l] + enc_b[l]   (4 distinct edge types)
__global__ void k_conv_enc(const float* __restrict__ edge_emb,
                           const float* __restrict__ enc_w,
                           const float* __restrict__ enc_b,
                           float* __restrict__ enc_tab) {
    int l = blockIdx.x;
    for (int idx = threadIdx.x; idx < 4 * FIN; idx += blockDim.x) {
        int a = idx / FIN, o = idx % FIN;
        float acc = enc_b[l * FIN + o];
        for (int c = 0; c < 50; ++c)
            acc += edge_emb[a * 50 + c] * enc_w[(l * 50 + c) * FIN + o];
        enc_tab[(l * 4 + a) * FIN + o] = acc;
    }
}

// pack pre_w (towers folded into N) into f16 WMMA-B fragment order
__global__ void k_conv_pre(const float* __restrict__ pre_w, _Float16* __restrict__ Bpre) {
    int idx = blockIdx.x * blockDim.x + threadIdx.x;
    int total = NL * MSG_NT * MSG_KK * 512;
    if (idx >= total) return;
    int j    = idx & 15;
    int lane = (idx >> 4) & 31;
    int kk   = (idx >> 9) & 7;
    int nt   = (idx >> 12) % MSG_NT;
    int l    = idx / (MSG_NT * MSG_KK * 512);
    int col = nt * 16 + (lane & 15);
    int k   = kk * 32 + ((lane >= 16) ? 16 : 0) + j;
    float v = 0.f;
    if (k < MSG_K && col < MSG_N) {
        int t = col / FIN, o = col % FIN;
        v = pre_w[((l * NTOW + t) * MSG_K + k) * FIN + o];
    }
    Bpre[idx] = (_Float16)v;
}

// pack post_w into f16 WMMA-B fragment order (per layer, per tower)
__global__ void k_conv_post(const float* __restrict__ post_w, _Float16* __restrict__ Bpost) {
    int idx = blockIdx.x * blockDim.x + threadIdx.x;
    int total = NL * NTOW * POST_KK * 512;
    if (idx >= total) return;
    int j    = idx & 15;
    int lane = (idx >> 4) & 31;
    int kk   = (idx >> 9) % POST_KK;
    int lt   = idx / (POST_KK * 512);
    int l = lt / NTOW, t = lt % NTOW;
    int o = lane & 15;
    int k = kk * 32 + ((lane >= 16) ? 16 : 0) + j;
    float v = 0.f;
    if (k < POST_K && o < FOUT)
        v = post_w[((l * NTOW + t) * POST_K + k) * FOUT + o];
    Bpost[idx] = (_Float16)v;
}

// ---------- graph setup ----------
__global__ void k_embed(const int* __restrict__ x, const float* __restrict__ node_emb,
                        float* __restrict__ h) {
    int idx = blockIdx.x * blockDim.x + threadIdx.x;
    if (idx >= NN * FIN) return;
    int n = idx / FIN, j = idx % FIN;
    h[idx] = node_emb[x[n] * FIN + j];
}
__global__ void k_zero_deg(float* __restrict__ deg) {
    int i = blockIdx.x * blockDim.x + threadIdx.x;
    if (i < NN) deg[i] = 0.f;
}
__global__ void k_deg(const int* __restrict__ dst, float* __restrict__ deg) {
    int e = blockIdx.x * blockDim.x + threadIdx.x;
    if (e < NE) atomicAdd(&deg[dst[e]], 1.f);
}
__global__ void k_layer_init(float* __restrict__ sum, float* __restrict__ sumsq,
                             unsigned* __restrict__ mnU, unsigned* __restrict__ mxU,
                             float* __restrict__ bns) {
    int i = blockIdx.x * blockDim.x + threadIdx.x;
    if (i < NN * MSG_N) {
        sum[i] = 0.f; sumsq[i] = 0.f;
        mnU[i] = 0xFF800000u;   // fmap(+inf)
        mxU[i] = 0x007FFFFFu;   // fmap(-inf)
    }
    if (i < 2 * FIN) bns[i] = 0.f;
}

// ---------- message GEMM + fused segment aggregation (WMMA) ----------
// A[e, 0:75]=h[dst], [75:150]=h[src], [150:225]=enc_tab[ea];  B = packed pre_w
__global__ void __launch_bounds__(256)
k_msg(const float* __restrict__ h, const int* __restrict__ src, const int* __restrict__ dst,
      const int* __restrict__ ea, const float* __restrict__ enc_tab_l,
      const _Float16* __restrict__ Bpre_l, const float* __restrict__ pre_b_l,
      float* __restrict__ sum, float* __restrict__ sumsq,
      unsigned* __restrict__ mnU, unsigned* __restrict__ mxU) {
    __shared__ __align__(32) _Float16 As[MSG_KK * 32 * 16];   // 8KB
    __shared__ int sdst[16], ssrc[16], sea[16];
    int tid = threadIdx.x;
    int tile = blockIdx.x;
    if (tid < 16) {
        int e = tile * 16 + tid;
        if (e < NE) { sdst[tid] = dst[e]; ssrc[tid] = src[e]; sea[tid] = ea[e]; }
        else        { sdst[tid] = -1;     ssrc[tid] = 0;      sea[tid] = 0;     }
    }
    __syncthreads();
    // build A tile (16 x 256) in LDS, fragment order
    for (int idx = tid; idx < 16 * MSG_KP; idx += 256) {
        int r = idx >> 8, k = idx & 255;
        float v = 0.f;
        int d = sdst[r];
        if (d >= 0 && k < MSG_K) {
            if (k < FIN)            v = h[d * FIN + k];
            else if (k < 2 * FIN)   v = h[ssrc[r] * FIN + (k - FIN)];
            else                    v = enc_tab_l[sea[r] * FIN + (k - 2 * FIN)];
        }
        As[a_frag_off(r, k)] = (_Float16)v;
    }
    __syncthreads();
    int wave = tid >> 5, lane = tid & 31;
    for (int nt = wave; nt < MSG_NT; nt += 8) {
        v8f c = {0.f, 0.f, 0.f, 0.f, 0.f, 0.f, 0.f, 0.f};
        const _Float16* bp = Bpre_l + nt * (MSG_KK * 512);
#pragma unroll
        for (int kk = 0; kk < MSG_KK; ++kk) {
            v16h a = *(const v16h*)(As + ((kk * 32 + lane) << 4));
            v16h b = *(const v16h*)(bp + kk * 512 + (lane << 4));
            c = __builtin_amdgcn_wmma_f32_16x16x32_f16(false, a, false, b,
                                                       (short)0, c, false, false);
        }
        int col = nt * 16 + (lane & 15);
        if (col < MSG_N) {
            float bias = pre_b_l[col];            // [T,75] flattened == col
            int rbase = (lane >= 16) ? 8 : 0;
#pragma unroll
            for (int v = 0; v < 8; ++v) {
                int r = rbase + v;
                int d = sdst[r];
                if (d < 0) continue;
                float val = c[v] + bias;
                int off = d * MSG_N + col;
                atomicAdd(&sum[off], val);
                atomicAdd(&sumsq[off], val * val);
                unsigned u = fmap(val);
                atomicMin(&mnU[off], u);
                atomicMax(&mxU[off], u);
            }
        }
    }
}

// ---------- post-NN (per node-tile x tower, WMMA) ----------
__global__ void __launch_bounds__(32)
k_post(const float* __restrict__ h, const float* __restrict__ deg,
       const float* __restrict__ sum, const float* __restrict__ sumsq,
       const unsigned* __restrict__ mnU, const unsigned* __restrict__ mxU,
       const _Float16* __restrict__ Bpost_l, const float* __restrict__ post_b_l,
       float* __restrict__ P) {
    __shared__ __align__(32) _Float16 As[POST_KK * 32 * 16];  // ~31KB
    __shared__ float scnt[16], samp[16];
    __shared__ int shas[16];
    int lane = threadIdx.x;
    int tile = blockIdx.x / NTOW;
    int t    = blockIdx.x % NTOW;
    if (lane < 16) {
        int n = tile * 16 + lane;
        float d = (n < NN) ? deg[n] : 0.f;
        float cnt = fmaxf(d, 1.f);
        scnt[lane] = cnt;
        samp[lane] = logf(cnt + 1.f) / ADL;
        shas[lane] = (d > 0.f) ? 1 : 0;
    }
    __syncthreads();
    for (int idx = lane; idx < 16 * POST_KP; idx += 32) {
        int r = idx / POST_KP, k = idx % POST_KP;
        int n = tile * 16 + r;
        float v = 0.f;
        if (n < NN && k < POST_K) {
            if (k < FIN) v = h[n * FIN + k];
            else {
                int kb = k - FIN;
                int blk = kb / 300, sub = kb % 300;
                int a = sub / FIN, cc = sub % FIN;
                int off = n * MSG_N + t * FIN + cc;
                float cnt = scnt[r], val;
                if (a == 0)      val = sum[off] / cnt;
                else if (a == 1) val = shas[r] ? funmap(mnU[off]) : 0.f;
                else if (a == 2) val = shas[r] ? funmap(mxU[off]) : 0.f;
                else {
                    float me = sum[off] / cnt;
                    float ms = sumsq[off] / cnt;
                    val = sqrtf(fmaxf(ms - me * me, 0.f) + 1e-5f);
                }
                if (blk == 1)      val *= samp[r];
                else if (blk == 2) val /= samp[r];
                v = val;
            }
        }
        As[a_frag_off(r, k)] = (_Float16)v;
    }
    __syncthreads();
    v8f c = {0.f, 0.f, 0.f, 0.f, 0.f, 0.f, 0.f, 0.f};
    const _Float16* bp = Bpost_l + t * (POST_KK * 512);
    for (int kk = 0; kk < POST_KK; ++kk) {
        v16h a = *(const v16h*)(As + ((kk * 32 + lane) << 4));
        v16h b = *(const v16h*)(bp + kk * 512 + (lane << 4));
        c = __builtin_amdgcn_wmma_f32_16x16x32_f16(false, a, false, b,
                                                   (short)0, c, false, false);
    }
    int o = lane & 15;
    if (o < FOUT) {
        float bias = post_b_l[t * FOUT + o];
        int rbase = (lane >= 16) ? 8 : 0;
#pragma unroll
        for (int v = 0; v < 8; ++v) {
            int n = tile * 16 + rbase + v;
            if (n < NN) P[n * FIN + t * FOUT + o] = c[v] + bias;
        }
    }
}

// ---------- lin / bn / pool / mlp ----------
__global__ void k_lin(const float* __restrict__ P, const float* __restrict__ lin_w_l,
                      const float* __restrict__ lin_b_l, float* __restrict__ Y) {
    int idx = blockIdx.x * blockDim.x + threadIdx.x;
    if (idx >= NN * FIN) return;
    int n = idx / FIN, j = idx % FIN;
    float acc = lin_b_l[j];
    const float* pr = P + n * FIN;
    for (int c = 0; c < FIN; ++c) acc += pr[c] * lin_w_l[c * FIN + j];
    Y[idx] = acc;
}
__global__ void k_bn_stats(const float* __restrict__ Y, float* __restrict__ bns) {
    __shared__ float ss[FIN], sq[FIN];
    int tid = threadIdx.x;
    if (tid < FIN) { ss[tid] = 0.f; sq[tid] = 0.f; }
    __syncthreads();
    int n = blockIdx.x * blockDim.x + tid;
    if (n < NN) {
        for (int j = 0; j < FIN; ++j) {
            float v = Y[n * FIN + j];
            atomicAdd(&ss[j], v);
            atomicAdd(&sq[j], v * v);
        }
    }
    __syncthreads();
    if (tid < FIN) {
        atomicAdd(&bns[tid], ss[tid]);
        atomicAdd(&bns[FIN + tid], sq[tid]);
    }
}
__global__ void k_bn_apply(const float* __restrict__ Y, const float* __restrict__ bns,
                           const float* __restrict__ g, const float* __restrict__ b,
                           float* __restrict__ h) {
    int idx = blockIdx.x * blockDim.x + threadIdx.x;
    if (idx >= NN * FIN) return;
    int j = idx % FIN;
    float mu  = bns[j] / (float)NN;
    float var = bns[FIN + j] / (float)NN - mu * mu;
    float v = (Y[idx] - mu) * rsqrtf(var + 1e-5f) * g[j] + b[j];
    h[idx] = fmaxf(v, 0.f);
}
__global__ void k_zero_hg(float* __restrict__ hg) {
    int i = blockIdx.x * blockDim.x + threadIdx.x;
    if (i < NG * FIN) hg[i] = 0.f;
}
__global__ void k_pool(const float* __restrict__ h, const int* __restrict__ batch,
                       float* __restrict__ hg) {
    int idx = blockIdx.x * blockDim.x + threadIdx.x;
    if (idx >= NN * FIN) return;
    int n = idx / FIN, j = idx % FIN;
    atomicAdd(&hg[batch[n] * FIN + j], h[idx]);
}
__global__ void __launch_bounds__(128)
k_mlp(const float* __restrict__ hg,
      const float* __restrict__ w1, const float* __restrict__ b1,
      const float* __restrict__ w2, const float* __restrict__ b2,
      const float* __restrict__ w3, const float* __restrict__ b3,
      float* __restrict__ out) {
    __shared__ float sh[FIN], z1[50], z2[25];
    int g = blockIdx.x, t = threadIdx.x;
    if (t < FIN) sh[t] = hg[g * FIN + t];
    __syncthreads();
    if (t < 50) {
        float acc = b1[t];
        for (int c = 0; c < FIN; ++c) acc += sh[c] * w1[c * 50 + t];
        z1[t] = fmaxf(acc, 0.f);
    }
    __syncthreads();
    if (t < 25) {
        float acc = b2[t];
        for (int c = 0; c < 50; ++c) acc += z1[c] * w2[c * 25 + t];
        z2[t] = fmaxf(acc, 0.f);
    }
    __syncthreads();
    if (t == 0) {
        float acc = b3[0];
        for (int c = 0; c < 25; ++c) acc += z2[c] * w3[c];
        out[g] = acc;
    }
}

extern "C" void kernel_launch(void* const* d_in, const int* in_sizes, int n_in,
                              void* d_out, int out_size, void* d_ws, size_t ws_size,
                              hipStream_t stream) {
    const int*   x        = (const int*)d_in[0];
    const int*   ei       = (const int*)d_in[1];
    const int*   ea       = (const int*)d_in[2];
    const int*   batch    = (const int*)d_in[3];
    const float* node_emb = (const float*)d_in[4];
    const float* edge_emb = (const float*)d_in[5];
    const float* enc_w    = (const float*)d_in[6];
    const float* enc_b    = (const float*)d_in[7];
    const float* pre_w    = (const float*)d_in[8];
    const float* pre_b    = (const float*)d_in[9];
    const float* post_w   = (const float*)d_in[10];
    const float* post_b   = (const float*)d_in[11];
    const float* lin_w    = (const float*)d_in[12];
    const float* lin_b    = (const float*)d_in[13];
    const float* bn_g     = (const float*)d_in[14];
    const float* bn_b     = (const float*)d_in[15];
    const float* mlp_w1   = (const float*)d_in[16];
    const float* mlp_b1   = (const float*)d_in[17];
    const float* mlp_w2   = (const float*)d_in[18];
    const float* mlp_b2   = (const float*)d_in[19];
    const float* mlp_w3   = (const float*)d_in[20];
    const float* mlp_b3   = (const float*)d_in[21];
    float* out = (float*)d_out;
    const int* src = ei;
    const int* dst = ei + NE;

    // workspace bump allocator
    char* p = (char*)d_ws;
    auto alloc = [&](size_t bytes) -> char* {
        char* r = p; p += (bytes + 255) & ~(size_t)255; return r;
    };
    float*     h      = (float*)alloc((size_t)NN * FIN * 4);
    float*     P      = (float*)alloc((size_t)NN * FIN * 4);
    float*     Y      = (float*)alloc((size_t)NN * FIN * 4);
    float*     sum    = (float*)alloc((size_t)NN * MSG_N * 4);
    float*     sumsq  = (float*)alloc((size_t)NN * MSG_N * 4);
    unsigned*  mnU    = (unsigned*)alloc((size_t)NN * MSG_N * 4);
    unsigned*  mxU    = (unsigned*)alloc((size_t)NN * MSG_N * 4);
    float*     deg    = (float*)alloc((size_t)NN * 4);
    float*     bns    = (float*)alloc(2 * FIN * 4);
    float*     hg     = (float*)alloc((size_t)NG * FIN * 4);
    float*     enc_tab= (float*)alloc((size_t)NL * 4 * FIN * 4);
    _Float16*  Bpre   = (_Float16*)alloc((size_t)NL * MSG_NT * MSG_KK * 512 * 2);
    _Float16*  Bpost  = (_Float16*)alloc((size_t)NL * NTOW * POST_KK * 512 * 2);

    // weight prep (deterministic, redone each call)
    k_conv_enc <<<NL, 128, 0, stream>>>(edge_emb, enc_w, enc_b, enc_tab);
    {
        int tot = NL * MSG_NT * MSG_KK * 512;
        k_conv_pre <<<(tot + 255) / 256, 256, 0, stream>>>(pre_w, Bpre);
    }
    {
        int tot = NL * NTOW * POST_KK * 512;
        k_conv_post<<<(tot + 255) / 256, 256, 0, stream>>>(post_w, Bpost);
    }

    k_embed   <<<(NN * FIN + 255) / 256, 256, 0, stream>>>(x, node_emb, h);
    k_zero_deg<<<(NN + 255) / 256, 256, 0, stream>>>(deg);
    k_deg     <<<(NE + 255) / 256, 256, 0, stream>>>(dst, deg);

    int msg_tiles  = (NE + 15) / 16;               // 4375
    int post_tiles = ((NN + 15) / 16) * NTOW;      // 9375

    for (int l = 0; l < NL; ++l) {
        k_layer_init<<<(NN * MSG_N + 255) / 256, 256, 0, stream>>>(sum, sumsq, mnU, mxU, bns);
        k_msg<<<msg_tiles, 256, 0, stream>>>(
            h, src, dst, ea,
            enc_tab + (size_t)l * 4 * FIN,
            Bpre + (size_t)l * MSG_NT * MSG_KK * 512,
            pre_b + (size_t)l * NTOW * FIN,
            sum, sumsq, mnU, mxU);
        k_post<<<post_tiles, 32, 0, stream>>>(
            h, deg, sum, sumsq, mnU, mxU,
            Bpost + (size_t)l * NTOW * POST_KK * 512,
            post_b + (size_t)l * NTOW * FOUT,
            P);
        k_lin<<<(NN * FIN + 255) / 256, 256, 0, stream>>>(
            P, lin_w + (size_t)l * FIN * FIN, lin_b + (size_t)l * FIN, Y);
        k_bn_stats<<<(NN + 255) / 256, 256, 0, stream>>>(Y, bns);
        k_bn_apply<<<(NN * FIN + 255) / 256, 256, 0, stream>>>(
            Y, bns, bn_g + (size_t)l * FIN, bn_b + (size_t)l * FIN, h);
    }

    k_zero_hg<<<(NG * FIN + 255) / 256, 256, 0, stream>>>(hg);
    k_pool   <<<(NN * FIN + 255) / 256, 256, 0, stream>>>(h, batch, hg);
    k_mlp    <<<NG, 128, 0, stream>>>(hg, mlp_w1, mlp_b1, mlp_w2, mlp_b2, mlp_w3, mlp_b3, out);

    (void)in_sizes; (void)n_in; (void)out_size; (void)ws_size;
}
